// ActorCriticGAT_45062796869928
// MI455X (gfx1250) — matrix operand order repaired and checked by reading, not compiled, not run on hardware
//
#include <hip/hip_runtime.h>
#include <hip/hip_bf16.h>
#include <math.h>

#define NN 20000
#define EE 320000
#define GG 64
#define INF_ 128
#define HIDF 256
#define HEADS1 4
#define EDGED 5
#define ACT 8
#define NEG_SLOPE 0.2f

typedef _Float16 v16h __attribute__((ext_vector_type(16)));
typedef _Float16 v8h  __attribute__((ext_vector_type(8)));
typedef float    v8f  __attribute__((ext_vector_type(8)));

// ---------------------------------------------------------------------------
// Utility: float atomic max via CAS
// ---------------------------------------------------------------------------
__device__ __forceinline__ void atomicMaxF(float* addr, float val) {
    unsigned int* ua = (unsigned int*)addr;
    unsigned int old = *ua;
    while (true) {
        float f = __uint_as_float(old);
        if (f >= val) break;
        unsigned int assumed = old;
        old = atomicCAS(ua, assumed, __float_as_uint(val));
        if (old == assumed) break;
    }
}

// pack two f32 -> two f16 halves -> single 32-bit LDS store
__device__ __forceinline__ void packstore2(_Float16* dst, float lo, float hi) {
    union { unsigned u; _Float16 h[2]; } p;
    p.h[0] = (_Float16)lo;
    p.h[1] = (_Float16)hi;
    *(unsigned*)dst = p.u;
}

// pack four f32 -> four f16 halves -> single 64-bit LDS store
__device__ __forceinline__ void packstore4(_Float16* dst, float4 v) {
    union { unsigned long long u; _Float16 h[4]; } p;
    p.h[0] = (_Float16)v.x;
    p.h[1] = (_Float16)v.y;
    p.h[2] = (_Float16)v.z;
    p.h[3] = (_Float16)v.w;
    *(unsigned long long*)dst = p.u;
}

// ---------------------------------------------------------------------------
// fill kernel (scratch init; harness poisons ws, we must zero/-inf ourselves)
// ---------------------------------------------------------------------------
__global__ void fill_kernel(float* __restrict__ p, float v, int n) {
    int i = blockIdx.x * 256 + threadIdx.x;
    if (i < n) p[i] = v;
}

// ---------------------------------------------------------------------------
// WMMA GEMM: C[M x 256] = A[M x K] @ W[K x 256]   (f32 in, f16 WMMA, f32 acc)
// block: 256 threads = 8 waves, tile 32(M) x 256(N): waves = 2 row x 4 col,
// each wave owns 16x64 (4 accumulators, A fragment reused 4x).
// W tile stored TRANSPOSED in LDS so B fragments are contiguous b128 loads;
// staging uses packed f16x2 (b32) / f16x4 (b64) LDS stores.
// grid: (M/32).  M,K multiples of 32, N = 256 exactly. No divergence.
// ---------------------------------------------------------------------------
__global__ void gemm_wmma_f16(const float* __restrict__ A,
                              const float* __restrict__ W,
                              float* __restrict__ C,
                              int M, int K) {
    // rows padded to 48 halves (96B): keeps 16B alignment for b128 loads
    __shared__ _Float16 As[32][48];     //  3.0 KB  (A tile, row-major)
    __shared__ _Float16 WsT[256][48];   // 24.0 KB  ([col][k] = transposed)

    const int tid  = threadIdx.x;
    const int wave = tid >> 5;
    const int lane = tid & 31;
    const int wrow = wave >> 2;         // 0..1
    const int wcol = wave & 3;          // 0..3
    const int m0   = blockIdx.x * 32;

    v8f acc[4] = {};

    // A fragment: lane<16 holds row=lane, K={0..7,16..23}; lane>=16: K={8..15,24..31}
    const int arow = wrow * 16 + (lane & 15);
    const int koff = (lane < 16) ? 0 : 8;
    // B fragment: lane<16 holds col, K={0..7,16..23}; lane>=16: K={8..15,24..31}
    const int kb   = (lane < 16) ? 0 : 8;

    // staging coordinates
    const int ar = tid >> 3;            // A: row 0..31
    const int ac = (tid & 7) * 4;       // A: 4 consecutive cols
    const int rr = tid >> 4;            // W: k-row pair 0..15 -> rows 2rr, 2rr+1
    const int cb = (tid & 15) * 16;     // W: 16 consecutive cols

    for (int kk = 0; kk < K; kk += 32) {
        // stage A tile: 32x32, one float4 load + one packed b64 LDS store
        {
            float4 av = *(const float4*)&A[(long)(m0 + ar) * K + kk + ac];
            packstore4(&As[ar][ac], av);
        }
        // stage W tile transposed: two k-rows x 16 cols per thread,
        // float4 global loads, packed f16x2 b32 LDS stores.
        {
            const float4* w0 = (const float4*)&W[(long)(kk + 2 * rr) * HIDF + cb];
            const float4* w1 = (const float4*)&W[(long)(kk + 2 * rr + 1) * HIDF + cb];
            #pragma unroll
            for (int q = 0; q < 4; ++q) {
                float4 a0 = w0[q];
                float4 a1 = w1[q];
                const int c = cb + q * 4;
                packstore2(&WsT[c + 0][2 * rr], a0.x, a1.x);
                packstore2(&WsT[c + 1][2 * rr], a0.y, a1.y);
                packstore2(&WsT[c + 2][2 * rr], a0.z, a1.z);
                packstore2(&WsT[c + 3][2 * rr], a0.w, a1.w);
            }
        }
        __syncthreads();

        // A fragment: two aligned 16B LDS loads
        v8h alo = *(const v8h*)&As[arow][koff];
        v8h ahi = *(const v8h*)&As[arow][16 + koff];
        v16h afrag = __builtin_shufflevector(alo, ahi,
            0, 1, 2, 3, 4, 5, 6, 7, 8, 9, 10, 11, 12, 13, 14, 15);

        #pragma unroll
        for (int t = 0; t < 4; ++t) {
            const int bcol = wcol * 64 + t * 16 + (lane & 15);
            v8h b0 = *(const v8h*)&WsT[bcol][kb];
            v8h b1 = *(const v8h*)&WsT[bcol][kb + 16];
            v16h bfrag = __builtin_shufflevector(b0, b1,
                0, 1, 2, 3, 4, 5, 6, 7, 8, 9, 10, 11, 12, 13, 14, 15);
            acc[t] = __builtin_amdgcn_wmma_f32_16x16x32_f16(
                false, afrag, false, bfrag, (short)0, acc[t], false, false);
        }
        __syncthreads();
    }

    // D layout: element r of v8f in lane l -> row r + 8*(l>=16), col = l&15
    const int rbase = m0 + wrow * 16 + ((lane < 16) ? 0 : 8);
    #pragma unroll
    for (int t = 0; t < 4; ++t) {
        const int colg = wcol * 64 + t * 16 + (lane & 15);
        #pragma unroll
        for (int r = 0; r < 8; ++r)
            C[(long)(rbase + r) * HIDF + colg] = acc[t][r];
    }
}

// ---------------------------------------------------------------------------
// Edge pass 1: logit[e,h] = att . leaky_relu(xl[src]+xr[dst]+edge_attr@We)
// plus segment-max into lmax[dst,h]. One wave per edge, 8 channels per lane.
// ---------------------------------------------------------------------------
__global__ void edge_logits(const float* __restrict__ xl,
                            const float* __restrict__ xr,
                            const float* __restrict__ eattr,
                            const float* __restrict__ We,
                            const float* __restrict__ att,
                            const int* __restrict__ ei,
                            float* __restrict__ logit,
                            float* __restrict__ lmax,
                            int H) {
    const int wave = threadIdx.x >> 5;
    const int lane = threadIdx.x & 31;
    const int e = blockIdx.x * 8 + wave;
    if (e >= EE) return;
    const int src = ei[e];
    const int dst = ei[EE + e];

    float ea[EDGED];
    #pragma unroll
    for (int d = 0; d < EDGED; ++d) ea[d] = eattr[(long)e * EDGED + d];

    const int c0 = lane * 8;
    float s = 0.f;
    #pragma unroll
    for (int j = 0; j < 8; ++j) {
        const int c = c0 + j;
        float ec = 0.f;
        #pragma unroll
        for (int d = 0; d < EDGED; ++d) ec = fmaf(ea[d], We[d * HIDF + c], ec);
        float m = xl[(long)src * HIDF + c] + xr[(long)dst * HIDF + c] + ec;
        m = (m > 0.f) ? m : NEG_SLOPE * m;
        s = fmaf(att[c], m, s);
    }
    const int lph = (HIDF / H) / 8;                 // lanes per head: 8 (H=4) or 32 (H=1)
    for (int off = 1; off < lph; off <<= 1) s += __shfl_xor(s, off, 32);
    if ((lane & (lph - 1)) == 0) {
        const int h = lane / lph;
        logit[(long)e * H + h] = s;
        atomicMaxF(&lmax[(long)dst * H + h], s);
    }
}

// ---------------------------------------------------------------------------
// Edge pass 2: unnorm = exp(logit - lmax[dst]); denom[dst] += unnorm
// (unnorm written over logit buffer)
// ---------------------------------------------------------------------------
__global__ void edge_softmax_sum(float* __restrict__ logit,
                                 const float* __restrict__ lmax,
                                 float* __restrict__ denom,
                                 const int* __restrict__ ei,
                                 int H) {
    const int i = blockIdx.x * 256 + threadIdx.x;
    if (i >= EE * H) return;
    const int e = i / H;
    const int h = i - e * H;
    const int dst = ei[EE + e];
    const float u = __expf(logit[i] - lmax[dst * H + h]);
    logit[i] = u;
    atomicAdd(&denom[dst * H + h], u);
}

// ---------------------------------------------------------------------------
// Edge pass 3: agg[dst] += alpha * xl[src]; one wave per edge.
// ---------------------------------------------------------------------------
__global__ void edge_aggregate(const float* __restrict__ xl,
                               const float* __restrict__ unnorm,
                               const float* __restrict__ denom,
                               const int* __restrict__ ei,
                               float* __restrict__ agg,
                               int H) {
    const int wave = threadIdx.x >> 5;
    const int lane = threadIdx.x & 31;
    const int e = blockIdx.x * 8 + wave;
    if (e >= EE) return;
    const int src = ei[e];
    const int dst = ei[EE + e];
    const int c0 = lane * 8;
    const int h = c0 / (HIDF / H);
    const float alpha = unnorm[(long)e * H + h] /
                        (denom[(long)dst * H + h] + 1e-16f);
    #pragma unroll
    for (int j = 0; j < 8; ++j) {
        const int c = c0 + j;
        atomicAdd(&agg[(long)dst * HIDF + c], alpha * xl[(long)src * HIDF + c]);
    }
}

// ---------------------------------------------------------------------------
// h = relu(agg + bias), in place
// ---------------------------------------------------------------------------
__global__ void bias_relu(float* __restrict__ h, const float* __restrict__ b) {
    const int i = blockIdx.x * 256 + threadIdx.x;
    if (i >= NN * HIDF) return;
    const float v = h[i] + b[i & (HIDF - 1)];
    h[i] = fmaxf(v, 0.f);
}

// ---------------------------------------------------------------------------
// Global mean pool (segment sums + counts)
// ---------------------------------------------------------------------------
__global__ void pool_sum(const float* __restrict__ h,
                         const int* __restrict__ batch,
                         float* __restrict__ sums) {
    const int i = blockIdx.x * 256 + threadIdx.x;
    if (i >= NN * HIDF) return;
    const int node = i >> 8;
    const int c = i & (HIDF - 1);
    atomicAdd(&sums[batch[node] * HIDF + c], h[i]);
}

__global__ void pool_cnt(const int* __restrict__ batch, float* __restrict__ cnt) {
    const int i = blockIdx.x * 256 + threadIdx.x;
    if (i >= NN) return;
    atomicAdd(&cnt[batch[i]], 1.0f);
}

// ---------------------------------------------------------------------------
// Heads: one block per graph; 9 block-reductions (8 logits + value)
// ---------------------------------------------------------------------------
__global__ void head_kernel(const float* __restrict__ sums,
                            const float* __restrict__ cnt,
                            const float* __restrict__ Wp,
                            const float* __restrict__ bp,
                            const float* __restrict__ Wv,
                            const float* __restrict__ bv,
                            float* __restrict__ out) {
    __shared__ float red[256];
    const int g = blockIdx.x;
    const int t = threadIdx.x;
    const float gc = sums[g * HIDF + t] / fmaxf(cnt[g], 1.0f);
    for (int k = 0; k < ACT + 1; ++k) {
        red[t] = (k < ACT) ? gc * Wp[t * ACT + k] : gc * Wv[t];
        __syncthreads();
        for (int s = 128; s > 0; s >>= 1) {
            if (t < s) red[t] += red[t + s];
            __syncthreads();
        }
        if (t == 0) out[g * (ACT + 1) + k] = red[0] + ((k < ACT) ? bp[k] : bv[0]);
        __syncthreads();
    }
}

// ---------------------------------------------------------------------------
// launcher
// ---------------------------------------------------------------------------
extern "C" void kernel_launch(void* const* d_in, const int* in_sizes, int n_in,
                              void* d_out, int out_size, void* d_ws, size_t ws_size,
                              hipStream_t stream) {
    (void)in_sizes; (void)n_in; (void)out_size; (void)ws_size;

    const float* x     = (const float*)d_in[0];
    const int*   ei    = (const int*)  d_in[1];
    const int*   batch = (const int*)  d_in[2];
    const float* eattr = (const float*)d_in[3];
    const float* W1l   = (const float*)d_in[4];
    const float* W1r   = (const float*)d_in[5];
    const float* W1e   = (const float*)d_in[6];
    const float* att1  = (const float*)d_in[7];
    const float* b1    = (const float*)d_in[8];
    const float* W2l   = (const float*)d_in[9];
    const float* W2r   = (const float*)d_in[10];
    const float* W2e   = (const float*)d_in[11];
    const float* att2  = (const float*)d_in[12];
    const float* b2    = (const float*)d_in[13];
    const float* Wp    = (const float*)d_in[14];
    const float* bp    = (const float*)d_in[15];
    const float* Wv    = (const float*)d_in[16];
    const float* bv    = (const float*)d_in[17];
    float* out = (float*)d_out;

    // workspace layout (floats)
    const long NB = (long)NN * HIDF;       // 5,120,000
    float* B0    = (float*)d_ws;           // xl (layer1 then layer2)
    float* B1    = B0 + NB;                // xr
    float* B2    = B1 + NB;                // agg1 -> h1
    float* B3    = B2 + NB;                // agg2 -> h2
    float* LOGIT = B3 + NB;                // E * Hmax (4)
    float* LMAX  = LOGIT + (long)EE * 4;   // N * 4
    float* DENOM = LMAX + (long)NN * 4;    // N * 4
    float* SUMS  = DENOM + (long)NN * 4;   // G * 256
    float* CNT   = SUMS + (long)GG * HIDF; // G

    const int NH = NN * HIDF;
    dim3 blk(256);

    // ============================ Layer 1 (H=4) ============================
    fill_kernel<<<(NN * 4 + 255) / 256, blk, 0, stream>>>(LMAX, -3.0e38f, NN * 4);
    fill_kernel<<<(NN * 4 + 255) / 256, blk, 0, stream>>>(DENOM, 0.f, NN * 4);
    fill_kernel<<<(NH + 255) / 256, blk, 0, stream>>>(B2, 0.f, NH);

    gemm_wmma_f16<<<NN / 32, blk, 0, stream>>>(x, W1l, B0, NN, INF_);
    gemm_wmma_f16<<<NN / 32, blk, 0, stream>>>(x, W1r, B1, NN, INF_);

    edge_logits<<<EE / 8, blk, 0, stream>>>(B0, B1, eattr, W1e, att1, ei, LOGIT, LMAX, HEADS1);
    edge_softmax_sum<<<(EE * HEADS1 + 255) / 256, blk, 0, stream>>>(LOGIT, LMAX, DENOM, ei, HEADS1);
    edge_aggregate<<<EE / 8, blk, 0, stream>>>(B0, LOGIT, DENOM, ei, B2, HEADS1);
    bias_relu<<<(NH + 255) / 256, blk, 0, stream>>>(B2, b1);        // B2 = h1

    // ============================ Layer 2 (H=1) ============================
    fill_kernel<<<(NN + 255) / 256, blk, 0, stream>>>(LMAX, -3.0e38f, NN);
    fill_kernel<<<(NN + 255) / 256, blk, 0, stream>>>(DENOM, 0.f, NN);
    fill_kernel<<<(NH + 255) / 256, blk, 0, stream>>>(B3, 0.f, NH);

    gemm_wmma_f16<<<NN / 32, blk, 0, stream>>>(B2, W2l, B0, NN, HIDF);
    gemm_wmma_f16<<<NN / 32, blk, 0, stream>>>(B2, W2r, B1, NN, HIDF);

    edge_logits<<<EE / 8, blk, 0, stream>>>(B0, B1, eattr, W2e, att2, ei, LOGIT, LMAX, 1);
    edge_softmax_sum<<<(EE + 255) / 256, blk, 0, stream>>>(LOGIT, LMAX, DENOM, ei, 1);
    edge_aggregate<<<EE / 8, blk, 0, stream>>>(B0, LOGIT, DENOM, ei, B3, 1);
    bias_relu<<<(NH + 255) / 256, blk, 0, stream>>>(B3, b2);        // B3 = h2

    // ========================= Pool + heads ================================
    fill_kernel<<<(GG * HIDF + 255) / 256, blk, 0, stream>>>(SUMS, 0.f, GG * HIDF);
    fill_kernel<<<1, blk, 0, stream>>>(CNT, 0.f, GG);
    pool_sum<<<(NH + 255) / 256, blk, 0, stream>>>(B3, batch, SUMS);
    pool_cnt<<<(NN + 255) / 256, blk, 0, stream>>>(batch, CNT);
    head_kernel<<<GG, blk, 0, stream>>>(SUMS, CNT, Wp, bp, Wv, bv, out);
}
